// Autoencoder_79577154060856
// MI455X (gfx1250) — compile-verified, matrix operand order checked by reading, not run on hardware
//
#include <hip/hip_runtime.h>
#include <hip/hip_bf16.h>
#include <math.h>

// ---------------------------------------------------------------------------
// CDNA5 (gfx1250) autoencoder: implicit-GEMM conv / deconv / GDN via
// v_wmma_f32_16x16x32_bf16 (wave32). Per-wave tile: 16 Cout x 32 pixels,
// K unrolled by 64 -> 4 WMMAs per barrier round. All spatial div/mod are
// shifts (power-of-two H,W), k-decomposition hoisted per K-iteration.
// ---------------------------------------------------------------------------

typedef __bf16 bf16_t;
typedef __attribute__((ext_vector_type(16))) __bf16 v16bf;
typedef __attribute__((ext_vector_type(8)))  float  v8f;

#define TILE_P 256      // pixels per block (8 waves x 32 pixels)
#define KSTEP  64       // K per iteration (2 WMMA K-slices)
#define LDSPAD 72       // padded row (bf16): 144B stride, 16B aligned, conflict-free

enum { MODE_CONV = 0, MODE_DECONV = 1, MODE_GDN = 2 };
enum { EPI_RELU = 1, EPI_ADDSKIP = 2, EPI_SIGMOID = 4, EPI_GDN = 8, EPI_GDN_INV = 16 };

__global__ __launch_bounds__(256)
void conv_gemm_wmma(const float* __restrict__ in,
                    const float* __restrict__ wgt,   // conv/deconv: OIHW; gdn: gamma[C,C]
                    const float* __restrict__ bias,  // conv: bias; gdn: beta
                    const float* __restrict__ skip,  // residual input / gdn x
                    float* __restrict__ out,
                    int Nb, int Cin, int Hin, int Win,
                    int Cout, int hwoShift, int woShift,
                    int stride, int mode, int epi)
{
    __shared__ bf16_t ldsA[16][LDSPAD];       // [m (Cout row)][k]
    __shared__ bf16_t ldsB[TILE_P][LDSPAD];   // [n (pixel)][k]

    const int tid  = threadIdx.x;
    const int lane = tid & 31;
    const int wave = tid >> 5;
    const int HWo  = 1 << hwoShift;
    const int hwoMask = HWo - 1;
    const int woMask  = (1 << woShift) - 1;
    const int Ptot = Nb << hwoShift;
    const int Ktot = (mode == MODE_GDN) ? Cin : Cin * 9;
    const int p0   = blockIdx.x * TILE_P;
    const int co0  = blockIdx.y * 16;
    const int HinWin = Hin * Win;

    // constant-per-thread K lane inside a tile (256 % 64 == 0)
    const int klA = tid & 63;

    v8f acc[2] = {};

    for (int k0 = 0; k0 < Ktot; k0 += KSTEP) {
        const int  k   = k0 + klA;
        const bool kok = (k < Ktot);
        // hoisted k -> (ci,kh,kw) decomposition (constant divisors -> mul/shift)
        int ci = 0, kh = 0, kw = 0;
        if (mode != MODE_GDN) {
            ci = k / 9;
            int r = k - ci * 9;
            kh = r / 3;
            kw = r - kh * 3;
        } else {
            ci = k;
        }

        // ---- stage A tile: wgt[co*Ktot + k] (holds for OIHW conv AND gamma) ----
        {
            const int mbase = tid >> 6;                 // 0..3
#pragma unroll
            for (int i = 0; i < 4; ++i) {
                int m  = mbase + 4 * i;
                int co = co0 + m;
                float v = (kok && co < Cout) ? wgt[co * Ktot + k] : 0.f;
                ldsA[m][klA] = (bf16_t)v;
            }
        }
        // prefetch next weight K-slice (global_prefetch_b8)
        if (k0 + KSTEP < Ktot)
            __builtin_prefetch(&wgt[co0 * Ktot + k + KSTEP], 0, 1);

        // ---- stage B tile: im2col / dilated gather / x^2 ----
        {
            const int nbase = tid >> 6;                 // 0..3, n = nbase + 4*i
#pragma unroll 4
            for (int i = 0; i < TILE_P / 4; ++i) {
                int n = nbase + 4 * i;
                int p = p0 + n;
                float v = 0.f;
                if (kok && p < Ptot) {
                    int b  = p >> hwoShift;
                    int hw = p & hwoMask;
                    int oh = hw >> woShift;
                    int ow = hw & woMask;
                    if (mode == MODE_GDN) {
                        float xv = in[((b * Cin + ci) << hwoShift) + hw];
                        v = xv * xv;
                    } else {
                        int ih, iw; bool ok;
                        if (mode == MODE_CONV) {
                            ih = oh * stride + kh - 1;
                            iw = ow * stride + kw - 1;
                            ok = (ih >= 0) & (ih < Hin) & (iw >= 0) & (iw < Win);
                        } else {  // lhs_dilation=2, pad(1,2): only even dilated taps hit x
                            int dh = oh + kh - 1, dw = ow + kw - 1;
                            ok = (dh >= 0) & (dh < 2 * Hin - 1) & ((dh & 1) == 0) &
                                 (dw >= 0) & (dw < 2 * Win - 1) & ((dw & 1) == 0);
                            ih = dh >> 1; iw = dw >> 1;
                        }
                        if (ok) v = in[(b * Cin + ci) * HinWin + ih * Win + iw];
                    }
                }
                ldsB[n][klA] = (bf16_t)v;
            }
        }
        __syncthreads();

        // ---- 4 WMMAs: 2 K-slices x 2 pixel sub-tiles, A-fragment reused ----
        // 16-bit A/B layout: lane<16 holds K chunks [0..7],[16..23] (+32 per slice)
        const int c0 = (lane >> 4) << 3;
        const int mrow = lane & 15;
#pragma unroll
        for (int ks = 0; ks < 2; ++ks) {
            v16bf af;
            {
                const bf16_t* Ar = &ldsA[mrow][ks << 5];
#pragma unroll
                for (int e = 0; e < 8; ++e) {
                    af[e]     = Ar[c0 + e];
                    af[e + 8] = Ar[c0 + 16 + e];
                }
            }
#pragma unroll
            for (int t = 0; t < 2; ++t) {
                v16bf bf;
                const bf16_t* Br = &ldsB[wave * 32 + t * 16 + mrow][ks << 5];
#pragma unroll
                for (int e = 0; e < 8; ++e) {
                    bf[e]     = Br[c0 + e];
                    bf[e + 8] = Br[c0 + 16 + e];
                }
                acc[t] = __builtin_amdgcn_wmma_f32_16x16x32_bf16(false, af, false, bf,
                                                                 (short)0, acc[t], false, false);
            }
        }
        __syncthreads();
    }

    // ---- epilogue: C/D layout => vgpr r, lane l -> M = r + 8*(l>>4), N = l&15 ----
    const int n = lane & 15;
#pragma unroll
    for (int t = 0; t < 2; ++t) {
        const int p = p0 + wave * 32 + t * 16 + n;
        if (p < Ptot) {
            const int b  = p >> hwoShift;
            const int hw = p & hwoMask;
#pragma unroll
            for (int r = 0; r < 8; ++r) {
                int m  = r + ((lane >> 4) << 3);
                int co = co0 + m;
                if (co < Cout) {
                    float v = acc[t][r];
                    int oidx = ((b * Cout + co) << hwoShift) + hw;
                    if (epi & (EPI_GDN | EPI_GDN_INV)) {
                        float xv  = skip[oidx];
                        float nrm = sqrtf(v + bias[co]);
                        v = (epi & EPI_GDN_INV) ? xv * nrm : xv / nrm;
                    } else {
                        v += bias[co];
                        if (epi & EPI_RELU)    v = fmaxf(v, 0.f);
                        if (epi & EPI_ADDSKIP) v += skip[oidx];
                        if (epi & EPI_SIGMOID) v = 1.f / (1.f + __expf(-v));
                    }
                    out[oidx] = v;
                }
            }
        }
    }
}

// ---------------------------------------------------------------------------
// Elementwise + entropy kernels
// ---------------------------------------------------------------------------

__global__ void binarize_kernel(const float* __restrict__ in, float* __restrict__ out, int n)
{
    int i = blockIdx.x * blockDim.x + threadIdx.x;
    if (i < n) {
        float x = in[i];
        out[i] = (x > 0.f) ? 1.f : ((x < 0.f) ? -1.f : 0.f);
    }
}

__global__ void stats_init_kernel(unsigned* stats)
{
    // layout: [8 x {min,max}] then [8 x 256] hist
    for (int i = threadIdx.x; i < 16 + 8 * 256; i += blockDim.x) {
        if (i < 16) stats[i] = (i & 1) ? 0u : 0x7F800000u;  // min=+inf bits, max=0
        else        stats[i] = 0u;
    }
}

__global__ void minmax_kernel(const float* __restrict__ x, unsigned* stats, int perImg)
{
    __shared__ unsigned smin[256], smax[256];
    const int img = blockIdx.y;
    const int i   = blockIdx.x * 256 + threadIdx.x;
    unsigned vmin = 0x7F800000u, vmax = 0u;
    if (i < perImg) {
        unsigned b = __float_as_uint(x[img * perImg + i]);  // sigmoid output > 0
        vmin = b; vmax = b;
    }
    smin[threadIdx.x] = vmin; smax[threadIdx.x] = vmax;
    __syncthreads();
    for (int off = 128; off > 0; off >>= 1) {
        if (threadIdx.x < off) {
            smin[threadIdx.x] = min(smin[threadIdx.x], smin[threadIdx.x + off]);
            smax[threadIdx.x] = max(smax[threadIdx.x], smax[threadIdx.x + off]);
        }
        __syncthreads();
    }
    if (threadIdx.x == 0) {
        atomicMin(&stats[img * 2 + 0], smin[0]);
        atomicMax(&stats[img * 2 + 1], smax[0]);
    }
}

__global__ void hist_kernel(const float* __restrict__ x, unsigned* stats, int perImg)
{
    __shared__ unsigned h[256];
    h[threadIdx.x] = 0u;
    __syncthreads();
    const int img = blockIdx.y;
    const float mn = __uint_as_float(stats[img * 2 + 0]);
    const float mx = __uint_as_float(stats[img * 2 + 1]);
    const float scale = (mx > mn) ? 256.f / (mx - mn) : 0.f;
    const int i = blockIdx.x * 256 + threadIdx.x;
    if (i < perImg) {
        float v = x[img * perImg + i];
        int idx = (int)floorf((v - mn) * scale);
        idx = min(max(idx, 0), 255);
        atomicAdd(&h[idx], 1u);
    }
    __syncthreads();
    if (h[threadIdx.x]) atomicAdd(&stats[16 + img * 256 + threadIdx.x], h[threadIdx.x]);
}

__global__ void entropy_kernel(const unsigned* __restrict__ stats, float* __restrict__ out, int perImg)
{
    __shared__ float red[256];
    const int t = threadIdx.x;
    float s = 0.f;
    for (int img = 0; img < 8; ++img) {
        float p = (float)stats[16 + img * 256 + t] / (float)perImg;
        s += -p * __log2f(p + 1e-6f);
    }
    red[t] = s;
    __syncthreads();
    for (int off = 128; off > 0; off >>= 1) {
        if (t < off) red[t] += red[t + off];
        __syncthreads();
    }
    if (t == 0) out[0] = red[0] / 8.0f;
}

// ---------------------------------------------------------------------------
// Host orchestration
// ---------------------------------------------------------------------------

static inline int ilog2i(int v) { int s = 0; while ((1 << s) < v) ++s; return s; }

static inline void gemm(hipStream_t s, int mode, int epi,
                        const float* in, const float* w, const float* bias, const float* skip,
                        float* out, int Cin, int Hin, int Win,
                        int Cout, int Hout, int Wout, int stride)
{
    const int P = 8 * Hout * Wout;
    dim3 grid((P + TILE_P - 1) / TILE_P, (Cout + 15) / 16);
    conv_gemm_wmma<<<grid, 256, 0, s>>>(in, w, bias, skip, out,
                                        8, Cin, Hin, Win, Cout,
                                        ilog2i(Hout * Wout), ilog2i(Wout),
                                        stride, mode, epi);
}

extern "C" void kernel_launch(void* const* d_in, const int* in_sizes, int n_in,
                              void* d_out, int out_size, void* d_ws, size_t ws_size,
                              hipStream_t stream)
{
    (void)in_sizes; (void)n_in; (void)out_size; (void)ws_size;
    const float* x = (const float*)d_in[0];
    // enc params at d_in[1..18], dec params at d_in[19..36] (dict order)
    #define EP(i) ((const float*)d_in[1 + (i)])
    #define DP(i) ((const float*)d_in[19 + (i)])

    const size_t BUF = 8ull * 64 * 256 * 256;            // 33,554,432 floats (largest tensor)
    float* A = (float*)d_ws;
    float* B = A + BUF;
    unsigned* stats = (unsigned*)(B + BUF);

    float* recon   = (float*)d_out;                      // [8,3,512,512]
    float* latent  = recon + 8ull * 3 * 512 * 512;       // [8,512,32,32]
    const int perImg = 3 * 512 * 512;

    // ---------------- encoder ----------------
    gemm(stream, MODE_CONV, 0,        x, EP(0),  EP(1),  nullptr, A,   3, 512, 512,  64, 256, 256, 2);
    gemm(stream, MODE_GDN,  EPI_GDN,  A, EP(3),  EP(2),  A,       B,  64, 256, 256,  64, 256, 256, 1);
    gemm(stream, MODE_CONV, 0,        B, EP(4),  EP(5),  nullptr, A,  64, 256, 256, 128, 128, 128, 2);
    gemm(stream, MODE_GDN,  EPI_GDN,  A, EP(7),  EP(6),  A,       B, 128, 128, 128, 128, 128, 128, 1);
    gemm(stream, MODE_CONV, 0,        B, EP(8),  EP(9),  nullptr, A, 128, 128, 128, 256,  64,  64, 2);
    gemm(stream, MODE_GDN,  EPI_GDN,  A, EP(11), EP(10), A,       B, 256,  64,  64, 256,  64,  64, 1);
    gemm(stream, MODE_CONV, 0,        B, EP(12), EP(13), nullptr, A, 256,  64,  64, 512,  32,  32, 2);
    // resblock (512ch)
    gemm(stream, MODE_CONV, EPI_RELU,    A, EP(14), EP(15), nullptr, B,      512, 32, 32, 512, 32, 32, 1);
    gemm(stream, MODE_CONV, EPI_ADDSKIP, B, EP(16), EP(17), A,       latent, 512, 32, 32, 512, 32, 32, 1);

    // binarize (straight-through forward = sign)
    {
        const int n = 8 * 512 * 32 * 32;
        binarize_kernel<<<(n + 255) / 256, 256, 0, stream>>>(latent, A, n);
    }

    // ---------------- decoder ----------------
    gemm(stream, MODE_DECONV, 0,            A, DP(0),  DP(1),  nullptr, B, 512,  32,  32, 256,  64,  64, 1);
    gemm(stream, MODE_GDN,    EPI_GDN_INV,  B, DP(3),  DP(2),  B,       A, 256,  64,  64, 256,  64,  64, 1);
    gemm(stream, MODE_DECONV, 0,            A, DP(4),  DP(5),  nullptr, B, 256,  64,  64, 128, 128, 128, 1);
    gemm(stream, MODE_GDN,    EPI_GDN_INV,  B, DP(7),  DP(6),  B,       A, 128, 128, 128, 128, 128, 128, 1);
    gemm(stream, MODE_DECONV, 0,            A, DP(8),  DP(9),  nullptr, B, 128, 128, 128,  64, 256, 256, 1);
    gemm(stream, MODE_GDN,    EPI_GDN_INV,  B, DP(11), DP(10), B,       A,  64, 256, 256,  64, 256, 256, 1);
    gemm(stream, MODE_DECONV, 0,            A, DP(12), DP(13), nullptr, B,  64, 256, 256,   3, 512, 512, 1);
    // resblock (3ch) + sigmoid
    gemm(stream, MODE_CONV, EPI_RELU,                  B, DP(14), DP(15), nullptr, A,     3, 512, 512, 3, 512, 512, 1);
    gemm(stream, MODE_CONV, EPI_ADDSKIP | EPI_SIGMOID, A, DP(16), DP(17), B,       recon, 3, 512, 512, 3, 512, 512, 1);

    // ---------------- entropy ----------------
    stats_init_kernel<<<1, 256, 0, stream>>>(stats);
    dim3 g((perImg + 255) / 256, 8);
    minmax_kernel<<<g, 256, 0, stream>>>(recon, stats, perImg);
    hist_kernel<<<g, 256, 0, stream>>>(recon, stats, perImg);
    entropy_kernel<<<1, 256, 0, stream>>>(stats, latent + 8ull * 512 * 32 * 32, perImg);

    #undef EP
    #undef DP
}
